// GIN_G_70961449664988
// MI455X (gfx1250) — matrix operand order repaired and checked by reading, not compiled, not run on hardware
//
#include <hip/hip_runtime.h>
#include <hip/hip_bf16.h>

typedef __attribute__((ext_vector_type(16))) _Float16 v16h;
typedef __attribute__((ext_vector_type(8)))  _Float16 v8h;
typedef __attribute__((ext_vector_type(8)))  float    v8f;

#define N_NODES  100000
#define N_EDGES  1600000
#define N_GRAPHS 1000
#define F_INF    16
#define DH       128
#define BN_EPS   1e-5f

// ---------------------------------------------------------------- utilities
__global__ void k_zero_f32(float* __restrict__ p, int n) {
    int i = blockIdx.x * blockDim.x + threadIdx.x;
    if (i < n) p[i] = 0.f;
}

__global__ void k_copy_f32v4(float4* __restrict__ dst, const float4* __restrict__ src, int n4) {
    int i = blockIdx.x * blockDim.x + threadIdx.x;
    if (i < n4) dst[i] = src[i];
}

// ------------------------------------------------ edge scatter: agg[dst] += h[src]
// one thread = one edge x 4 features (float4 gather, 4 f32 atomics)
template<int LOG2D>
__global__ void k_edge_scatter(float* __restrict__ agg, const float* __restrict__ h,
                               const int* __restrict__ ei, long total4) {
    long i = (long)blockIdx.x * blockDim.x + threadIdx.x;
    if (i >= total4) return;
    const int Q = 1 << (LOG2D - 2);          // feature-quads per edge
    int e  = (int)(i >> (LOG2D - 2));
    int fq = (int)(i & (Q - 1));
    int s = ei[e];
    int d = ei[N_EDGES + e];
    float4 v = *(const float4*)(h + ((long)s << LOG2D) + fq * 4);
    float* a = agg + ((long)d << LOG2D) + fq * 4;
    atomicAdd(a + 0, v.x);
    atomicAdd(a + 1, v.y);
    atomicAdd(a + 2, v.z);
    atomicAdd(a + 3, v.w);
}

// ---------------------------------------------------------------- helpers
__device__ __forceinline__ v8h cvt8(const float* __restrict__ p) {
    float4 a = ((const float4*)p)[0];
    float4 b = ((const float4*)p)[1];
    v8h r;
    r[0] = (_Float16)a.x; r[1] = (_Float16)a.y; r[2] = (_Float16)a.z; r[3] = (_Float16)a.w;
    r[4] = (_Float16)b.x; r[5] = (_Float16)b.y; r[6] = (_Float16)b.z; r[7] = (_Float16)b.w;
    return r;
}

__device__ __forceinline__ v16h cat16(v8h lo, v8h hi) {
    return __builtin_shufflevector(lo, hi, 0,1,2,3,4,5,6,7,8,9,10,11,12,13,14,15);
}

// Stage W (f32 [KIN, DH]) into LDS pre-swizzled into per-lane WMMA-B fragments:
// ldsB[((kk/32)*8 + t)*512 + lane*16 + j] = f16(W[(kk + (lane>>4)*8 + (j&7) + (j>>3)*16)*DH + t*16 + (lane&15)])
template<int KIN, int KPAD>
__device__ __forceinline__ void stage_B(_Float16* __restrict__ ldsB,
                                        const float* __restrict__ W, int tid) {
    const int total = KPAD * DH;             // halves
    for (int i = tid; i < total; i += 256) {
        int j    = i & 15;
        int lane = (i >> 4) & 31;
        int t    = (i >> 9) & 7;
        int ks   = i >> 12;
        int k = ks * 32 + ((lane >> 4) << 3) + (j & 7) + ((j >> 3) << 4);
        int n = t * 16 + (lane & 15);
        float w = (k < KIN) ? W[k * DH + n] : 0.f;
        ldsB[i] = (_Float16)w;
    }
}

// ---------------------------------------------------------------- WMMA GEMM A
// z = relu( bn( s @ W + bias ) ), s: f32 [N, KIN], z: f16 [N, DH]
template<int KIN, int KPAD>
__global__ void __launch_bounds__(256)
k_gin_lin_bn_relu(const float* __restrict__ S,
                  const float* __restrict__ W,
                  const float* __restrict__ bias,
                  const float* __restrict__ gamma,
                  const float* __restrict__ beta,
                  _Float16* __restrict__ Z) {
    __shared__ _Float16 ldsB[KPAD * DH];
    const int tid = threadIdx.x;
    stage_B<KIN, KPAD>(ldsB, W, tid);
    __syncthreads();

    const int lane = tid & 31;
    const int wave = tid >> 5;
    const int hi   = lane >> 4;
    const int m    = lane & 15;
    const int row0 = blockIdx.x * 128 + wave * 16;

    v8f zero = {};
    v8f acc[8];
#pragma unroll
    for (int t = 0; t < 8; ++t) acc[t] = zero;

    const int arow = row0 + m;
    const int srow = arow < N_NODES ? arow : 0;   // clamp: rows >= N never stored

    for (int kk = 0; kk < KPAD; kk += 32) {
        v8h lo = cvt8(S + srow * KIN + kk + 8 * hi);
        v8h hi8;
        if constexpr (KIN == KPAD) {
            hi8 = cvt8(S + srow * KIN + kk + 16 + 8 * hi);
        } else {
            v8h z = {};            // K >= KIN: compile-time zero padding
            hi8 = z;
        }
        v16h a = cat16(lo, hi8);
        const _Float16* bp = ldsB + (kk >> 5) * 4096 + lane * 16;
#pragma unroll
        for (int t = 0; t < 8; ++t) {
            v8h b0 = *(const v8h*)(bp + t * 512);
            v8h b1 = *(const v8h*)(bp + t * 512 + 8);
            v16h b = cat16(b0, b1);
            acc[t] = __builtin_amdgcn_wmma_f32_16x16x32_f16(
                false, a, false, b, (short)0, acc[t], false, false);
        }
    }

    const float rs = rsqrtf(1.f + BN_EPS);
    float al[8], b2[8];
#pragma unroll
    for (int t = 0; t < 8; ++t) {
        int col = t * 16 + m;
        al[t] = gamma[col] * rs;
        b2[t] = bias[col] * al[t] + beta[col];
    }
#pragma unroll
    for (int i = 0; i < 8; ++i) {
        int row = row0 + (hi << 3) + i;
        if (row < N_NODES) {
#pragma unroll
            for (int t = 0; t < 8; ++t) {
                float v = acc[t][i] * al[t] + b2[t];
                v = v > 0.f ? v : 0.f;
                Z[row * DH + t * 16 + m] = (_Float16)v;
            }
        }
    }
}

// ---------------------------------------------------------------- WMMA GEMM B
// h = relu( z @ W + bias ); optionally store h (f32) and pool[batch[row]] += h
__global__ void __launch_bounds__(256)
k_gin_lin_relu_pool(const _Float16* __restrict__ Z,
                    const float* __restrict__ W,       // [DH, DH]
                    const float* __restrict__ bias,
                    float* __restrict__ H,             // may be nullptr
                    float* __restrict__ pool,          // [G, DH]
                    const int* __restrict__ batch) {
    __shared__ _Float16 ldsB[DH * DH];
    const int tid = threadIdx.x;
    stage_B<DH, DH>(ldsB, W, tid);
    __syncthreads();

    const int lane = tid & 31;
    const int wave = tid >> 5;
    const int hi   = lane >> 4;
    const int m    = lane & 15;
    const int row0 = blockIdx.x * 128 + wave * 16;

    v8f zero = {};
    v8f acc[8];
#pragma unroll
    for (int t = 0; t < 8; ++t) acc[t] = zero;

    const int arow = row0 + m;
    const int srow = arow < N_NODES ? arow : 0;

    for (int kk = 0; kk < DH; kk += 32) {
        const _Float16* ap = Z + srow * DH + kk + 8 * hi;
        v8h lo  = *(const v8h*)(ap);
        v8h hi8 = *(const v8h*)(ap + 16);
        v16h a = cat16(lo, hi8);
        const _Float16* bp = ldsB + (kk >> 5) * 4096 + lane * 16;
#pragma unroll
        for (int t = 0; t < 8; ++t) {
            v8h b0 = *(const v8h*)(bp + t * 512);
            v8h b1 = *(const v8h*)(bp + t * 512 + 8);
            v16h b = cat16(b0, b1);
            acc[t] = __builtin_amdgcn_wmma_f32_16x16x32_f16(
                false, a, false, b, (short)0, acc[t], false, false);
        }
    }

    float bb[8];
#pragma unroll
    for (int t = 0; t < 8; ++t) bb[t] = bias[t * 16 + m];

#pragma unroll
    for (int i = 0; i < 8; ++i) {
        int row = row0 + (hi << 3) + i;
        if (row < N_NODES) {
            int g = batch[row];
            float* pp = pool + g * DH + m;
#pragma unroll
            for (int t = 0; t < 8; ++t) {
                float v = acc[t][i] + bb[t];
                v = v > 0.f ? v : 0.f;
                if (H) H[row * DH + t * 16 + m] = v;
                atomicAdd(pp + t * 16, v);
            }
        }
    }
}

// ---------------------------------------------------------------- readout head
__global__ void __launch_bounds__(256)
k_head(const float* __restrict__ p1, const float* __restrict__ p2,
       const float* __restrict__ stats,
       const float* __restrict__ wl1, const float* __restrict__ bl1,  // [266,384],[384]
       const float* __restrict__ wl2, const float* __restrict__ bl2,  // [384,5],[5]
       float* __restrict__ out) {
    const int FI  = 2 * DH + 10;  // 266
    const int HID = 3 * DH;       // 384
    __shared__ float feat[2 * DH + 10];
    __shared__ float hid[3 * DH];
    __shared__ float logit[5];
    __shared__ float red;
    const int g = blockIdx.x, tid = threadIdx.x;

    for (int i = tid; i < FI; i += 256)
        feat[i] = (i < DH) ? p1[g * DH + i]
                : (i < 2 * DH) ? p2[g * DH + i - DH]
                : stats[g * 10 + i - 2 * DH];
    __syncthreads();

    for (int j = tid; j < HID; j += 256) {
        float a = bl1[j];
        for (int k = 0; k < FI; ++k) a += feat[k] * wl1[k * HID + j];
        hid[j] = a > 0.f ? a : 0.f;
    }
    __syncthreads();

    if (tid < 5) {
        float a = bl2[tid];
        for (int k = 0; k < HID; ++k) a += hid[k] * wl2[k * 5 + tid];
        logit[tid] = a;
    }
    __syncthreads();
    if (tid == 0) {
        float mx = logit[0];
        for (int j = 1; j < 5; ++j) mx = fmaxf(mx, logit[j]);
        float s = 0.f;
        for (int j = 0; j < 5; ++j) s += expf(logit[j] - mx);
        red = mx + logf(s);
    }
    __syncthreads();
    if (tid < 5) out[g * 5 + tid] = logit[tid] - red;
}

// ---------------------------------------------------------------- launch
extern "C" void kernel_launch(void* const* d_in, const int* in_sizes, int n_in,
                              void* d_out, int out_size, void* d_ws, size_t ws_size,
                              hipStream_t stream) {
    (void)in_sizes; (void)n_in; (void)out_size; (void)ws_size;
    const float* x     = (const float*)d_in[0];
    const int*   ei    = (const int*)d_in[1];
    const int*   batch = (const int*)d_in[2];
    /* d_in[3] = eig: unused by the reference network */
    const float* stats = (const float*)d_in[4];
    const float* w1a = (const float*)d_in[5];
    const float* b1a = (const float*)d_in[6];
    const float* g1  = (const float*)d_in[7];
    const float* be1 = (const float*)d_in[8];
    const float* w1b = (const float*)d_in[9];
    const float* b1b = (const float*)d_in[10];
    const float* w2a = (const float*)d_in[11];
    const float* b2a = (const float*)d_in[12];
    const float* g2  = (const float*)d_in[13];
    const float* be2 = (const float*)d_in[14];
    const float* w2b = (const float*)d_in[15];
    const float* b2b = (const float*)d_in[16];
    const float* wl1 = (const float*)d_in[17];
    const float* bl1 = (const float*)d_in[18];
    const float* wl2 = (const float*)d_in[19];
    const float* bl2 = (const float*)d_in[20];
    float* out = (float*)d_out;

    char* ws = (char*)d_ws;
    size_t off = 0;
    auto carve = [&](size_t bytes) -> void* {
        void* p = ws + off;
        off += (bytes + 255) & ~(size_t)255;
        return p;
    };
    float*    agg = (float*)   carve((size_t)N_NODES * DH * sizeof(float));   // 51.2 MB
    float*    h1  = (float*)   carve((size_t)N_NODES * DH * sizeof(float));   // 51.2 MB
    _Float16* zb  = (_Float16*)carve((size_t)N_NODES * DH * sizeof(_Float16));// 25.6 MB
    float*    p1  = (float*)   carve((size_t)N_GRAPHS * DH * sizeof(float));
    float*    p2  = (float*)   carve((size_t)N_GRAPHS * DH * sizeof(float));

    const int nGemmBlocks = (N_NODES + 127) / 128;

    // pools accumulate via atomics -> must be zeroed each call
    k_zero_f32<<<(N_GRAPHS * DH + 255) / 256, 256, 0, stream>>>(p1, N_GRAPHS * DH);
    k_zero_f32<<<(N_GRAPHS * DH + 255) / 256, 256, 0, stream>>>(p2, N_GRAPHS * DH);

    // ---- layer 1 ----
    {
        const int n4 = N_NODES * F_INF / 4;
        k_copy_f32v4<<<(n4 + 255) / 256, 256, 0, stream>>>((float4*)agg, (const float4*)x, n4);
        const long tot4 = (long)N_EDGES * (F_INF / 4);
        k_edge_scatter<4><<<(int)((tot4 + 255) / 256), 256, 0, stream>>>(agg, x, ei, tot4);
    }
    k_gin_lin_bn_relu<F_INF, 32><<<nGemmBlocks, 256, 0, stream>>>(agg, w1a, b1a, g1, be1, zb);
    k_gin_lin_relu_pool<<<nGemmBlocks, 256, 0, stream>>>(zb, w1b, b1b, h1, p1, batch);

    // ---- layer 2 ----
    {
        const int n4 = N_NODES * DH / 4;
        k_copy_f32v4<<<(n4 + 255) / 256, 256, 0, stream>>>((float4*)agg, (const float4*)h1, n4);
        const long tot4 = (long)N_EDGES * (DH / 4);
        k_edge_scatter<7><<<(int)((tot4 + 255) / 256), 256, 0, stream>>>(agg, h1, ei, tot4);
    }
    k_gin_lin_bn_relu<DH, DH><<<nGemmBlocks, 256, 0, stream>>>(agg, w2a, b2a, g2, be2, zb);
    k_gin_lin_relu_pool<<<nGemmBlocks, 256, 0, stream>>>(zb, w2b, b2b, nullptr, p2, batch);

    // ---- readout ----
    k_head<<<N_GRAPHS, 256, 0, stream>>>(p1, p2, stats, wl1, bl1, wl2, bl2, out);
}
